// ConvFormer_42915313221976
// MI455X (gfx1250) — compile-verified, matrix operand order checked by reading, not run on hardware
//
#include <hip/hip_runtime.h>
#include <hip/hip_bf16.h>
#include <math.h>
#include <stdint.h>

// ---------------------------------------------------------------------------
// Types / helpers
// ---------------------------------------------------------------------------
typedef unsigned short bf16_t;
typedef __attribute__((ext_vector_type(16))) __bf16 v16bf;
typedef __attribute__((ext_vector_type(8)))  float  v8f;

#define BDIM 8
#define CDIM 256
#define NPIX 1024       // 32*32
#define OC   512        // NH*KQV
#define NH   8
#define HD   64
#define KIM  2304       // 9*256 (tap-major, channel-minor)

__device__ __forceinline__ bf16_t f2bf(float f) {
    unsigned u = __float_as_uint(f);
    unsigned r = u + 0x7fffu + ((u >> 16) & 1u);   // round-to-nearest-even
    return (bf16_t)(r >> 16);
}

// CDNA5 async copy: 16 bytes global -> LDS, tracked by ASYNCcnt.
// Generic LDS pointers keep the LDS byte address in the low 32 bits.
__device__ __forceinline__ void async_copy16(unsigned lds_off, const void* gptr) {
    asm volatile("global_load_async_to_lds_b128 %0, %1, off"
                 :: "v"(lds_off), "v"((unsigned long long)(uintptr_t)gptr)
                 : "memory");
}
__device__ __forceinline__ void wait_async() {
    asm volatile("s_wait_asynccnt 0x0" ::: "memory");
}

// Load one 16x32 bf16 WMMA operand fragment from an LDS tile stored K-major
// (row = M or N index, 32 contiguous K elements per row, 4-byte aligned).
// Layout per CDNA5 ISA 7.12.2: VGPR r holds K = {2r,2r+1} (r<4) or {16+2(r-4),..},
// +8 for lanes 16..31.
__device__ __forceinline__ v16bf load_frag_kmajor(const bf16_t* rowptr, int half) {
    union { v16bf v; unsigned u[8]; } r;
#pragma unroll
    for (int i = 0; i < 8; ++i) {
        int kb = ((i < 4) ? (2 * i) : (16 + 2 * (i - 4))) + 8 * half;
        r.u[i] = *reinterpret_cast<const unsigned*>(rowptr + kb);
    }
    return r.v;
}

__device__ __forceinline__ v8f wmma_bf16(v16bf a, v16bf b, v8f c) {
    return __builtin_amdgcn_wmma_f32_16x16x32_bf16(false, a, false, b, (short)0, c,
                                                   false, false);
}

// ---------------------------------------------------------------------------
// K0a: f32 -> bf16 convert (plain layout)
// ---------------------------------------------------------------------------
__global__ __launch_bounds__(256) void convert_kernel(const float* __restrict__ src,
                                                      bf16_t* __restrict__ dst, int n) {
    int i = blockIdx.x * 256 + threadIdx.x;
    if (i < n) dst[i] = f2bf(src[i]);
}

// ---------------------------------------------------------------------------
// K0b: conv weight permute [OC][C][3][3] -> [OC][9][C] bf16 (tap-major K order,
// so a 32-wide K chunk has a uniform tap and contiguous channels).
// ---------------------------------------------------------------------------
__global__ __launch_bounds__(256) void permute_wconv_kernel(const float* __restrict__ src,
                                                            bf16_t* __restrict__ dst) {
    int i = blockIdx.x * 256 + threadIdx.x;        // over OC*KIM
    int oc = i / KIM, r = i - oc * KIM;            // r = tap*256 + c
    int tap = r >> 8, c = r & 255;
    dst[i] = f2bf(src[(size_t)oc * KIM + c * 9 + tap]);
}

// ---------------------------------------------------------------------------
// K1: xp = x + pos_encoding_2d  (f32 + bf16 copies)
// ---------------------------------------------------------------------------
__global__ __launch_bounds__(256) void pe_add_kernel(const float* __restrict__ x,
                                                     float* __restrict__ xpF,
                                                     bf16_t* __restrict__ xpB) {
    int idx = blockIdx.x * 256 + threadIdx.x;      // [B][C][N]
    int n = idx & (NPIX - 1);
    int c = (idx >> 10) & (CDIM - 1);
    int yy = n >> 5, xx = n & 31;
    float pos = (c < 128) ? (float)xx : (float)yy;
    int cc = (c < 128) ? c : c - 128;
    int j = cc >> 1;
    float div = __expf(-(logf(10000.0f) / 128.0f) * (float)(2 * j));
    float arg = pos * div;
    float pev = (cc & 1) ? __cosf(arg) : __sinf(arg);
    float v = x[idx] + pev;
    xpF[idx] = v;
    xpB[idx] = f2bf(v);
}

// ---------------------------------------------------------------------------
// K2: 3x3 SAME conv as implicit GEMM, tap-major K, double-buffered LDS,
// async weight staging.  out[b][oc][n].
// grid: (NPIX/64, OC/128, B), block 256 (8 waves -> 4x2 grid of 32x32 subtiles)
// ---------------------------------------------------------------------------
__global__ __launch_bounds__(256) void conv_gemm_kernel(const bf16_t* __restrict__ Wmat,
                                                        const float*  __restrict__ bias,
                                                        const bf16_t* __restrict__ xp,
                                                        bf16_t* __restrict__ out) {
    __shared__ bf16_t As[2][128][32];   // [oc][k]
    __shared__ bf16_t Bs[2][64][32];    // [n][k]
    const int t = threadIdx.x, wave = t >> 5, lane = t & 31;
    const int half = lane >> 4, lrow = lane & 15;
    const int ocblk = (wave & 3) * 32, nblk = (wave >> 2) * 32;
    const int nt = blockIdx.x * 64, octile = blockIdx.y * 128, b = blockIdx.z;
    const size_t xbase = (size_t)b * CDIM * NPIX;

    const int a_oc = t >> 1, a_seg = (t & 1) * 16;
    const int b_n = t >> 2, b_kseg = (t & 3) * 8;
    const int py = (nt + b_n) >> 5, px = (nt + b_n) & 31;

    auto loadA = [&](int k0, int buf) {
        const bf16_t* src = Wmat + (size_t)(octile + a_oc) * KIM + k0 + a_seg;
        unsigned lo = (unsigned)(uintptr_t)&As[buf][a_oc][a_seg];
        async_copy16(lo, src);
        async_copy16(lo + 16, src + 8);
    };
    auto loadB = [&](int k0, int buf) {
        int tap = k0 >> 8;                       // uniform within a K chunk
        int ky = tap / 3, kx = tap - ky * 3;
        int yy = py + ky - 1, xx = px + kx - 1;
        bool inb = ((unsigned)yy < 32u) && ((unsigned)xx < 32u);
        const bf16_t* sp = xp + xbase + (size_t)((k0 & 255) + b_kseg) * NPIX
                           + yy * 32 + xx;
#pragma unroll
        for (int i = 0; i < 8; ++i)
            Bs[buf][b_n][b_kseg + i] = inb ? sp[(size_t)i * NPIX] : (bf16_t)0;
    };

    v8f acc[2][2] = {};
    loadA(0, 0); loadB(0, 0);
    wait_async();
    __syncthreads();
    int cur = 0;
    for (int k0 = 0; k0 < KIM; k0 += 32) {
        int nxt = cur ^ 1;
        if (k0 + 32 < KIM) { loadA(k0 + 32, nxt); loadB(k0 + 32, nxt); }
        v16bf a0 = load_frag_kmajor(&As[cur][ocblk + lrow][0], half);
        v16bf a1 = load_frag_kmajor(&As[cur][ocblk + 16 + lrow][0], half);
        v16bf b0 = load_frag_kmajor(&Bs[cur][nblk + lrow][0], half);
        v16bf b1 = load_frag_kmajor(&Bs[cur][nblk + 16 + lrow][0], half);
        acc[0][0] = wmma_bf16(a0, b0, acc[0][0]);
        acc[0][1] = wmma_bf16(a0, b1, acc[0][1]);
        acc[1][0] = wmma_bf16(a1, b0, acc[1][0]);
        acc[1][1] = wmma_bf16(a1, b1, acc[1][1]);
        wait_async();
        __syncthreads();
        cur = nxt;
    }
    // epilogue: + bias, store bf16 [b][oc][n]
#pragma unroll
    for (int i = 0; i < 2; ++i)
#pragma unroll
        for (int j = 0; j < 2; ++j)
#pragma unroll
            for (int r = 0; r < 8; ++r) {
                int oc_g = octile + ocblk + i * 16 + r + 8 * half;
                int n_g = nt + nblk + j * 16 + lrow;
                float v = acc[i][j][r] + bias[oc_g];
                out[(size_t)b * OC * NPIX + (size_t)oc_g * NPIX + n_g] = f2bf(v);
            }
}

// ---------------------------------------------------------------------------
// K3: fused attention. scores (K^T Q / N) -> softmax over HEADS -> att accum.
// grid: (NPIX/16, B), block 256; wave w owns head w; scores never hit memory.
// att stored [b][n][c*8+h] bf16 (reference's channel-major/head-minor layout).
// ---------------------------------------------------------------------------
__global__ __launch_bounds__(256) void attention_kernel(const bf16_t* __restrict__ kmat,
                                                        const bf16_t* __restrict__ qmat,
                                                        const bf16_t* __restrict__ vmat,
                                                        bf16_t* __restrict__ att) {
    __shared__ bf16_t Ks[NH][16][64];   // [h][n][c]  K-major for score A
    __shared__ bf16_t Qs[NH][32][64];   // [h][m][c]  K-major for score B
    __shared__ bf16_t Vs[NH][64][32];   // [h][c][m]  K-major for att  B
    __shared__ float  Ss[NH][16][32];   // raw scores
    __shared__ bf16_t Ms[NH][16][32];   // softmax weights, K-major for att A

    const int t = threadIdx.x, h = t >> 5, lane = t & 31;
    const int half = lane >> 4, lrow = lane & 15;
    const int nt = blockIdx.x * 16, b = blockIdx.y;
    const size_t base = (size_t)b * OC * NPIX;

    // stage K tile [h][c][nt..nt+15] -> Ks[h][n][c]
#pragma unroll
    for (int rr = 0; rr < 2; ++rr) {
        int oc = t * 2 + rr, hh = oc >> 6, c = oc & 63;
        const bf16_t* src = kmat + base + (size_t)oc * NPIX + nt;
#pragma unroll
        for (int i = 0; i < 16; ++i) Ks[hh][i][c] = src[i];
    }
    __syncthreads();

    v16bf ka0 = load_frag_kmajor(&Ks[h][lrow][0], half);    // c 0..31
    v16bf ka1 = load_frag_kmajor(&Ks[h][lrow][32], half);   // c 32..63
    v8f accv[4] = {};

    for (int mc = 0; mc < NPIX; mc += 32) {
        // stage Q (scattered) and V (async, contiguous) tiles for this m-chunk
#pragma unroll
        for (int rr = 0; rr < 2; ++rr) {
            int oc = t * 2 + rr, hh = oc >> 6, c = oc & 63;
            const bf16_t* qs = qmat + base + (size_t)oc * NPIX + mc;
#pragma unroll
            for (int m = 0; m < 32; ++m) Qs[hh][m][c] = qs[m];
            const bf16_t* vs = vmat + base + (size_t)oc * NPIX + mc;
            unsigned vlo = (unsigned)(uintptr_t)&Vs[hh][c][0];
#pragma unroll
            for (int j = 0; j < 4; ++j) async_copy16(vlo + j * 16, vs + j * 8);
        }
        wait_async();
        __syncthreads();

        // scores for my head: s[n][m] = sum_c k[c][n] q[c][m], scaled by 1/N
#pragma unroll
        for (int mt = 0; mt < 2; ++mt) {
            v16bf qb0 = load_frag_kmajor(&Qs[h][mt * 16 + lrow][0], half);
            v16bf qb1 = load_frag_kmajor(&Qs[h][mt * 16 + lrow][32], half);
            v8f s = {};
            s = wmma_bf16(ka0, qb0, s);
            s = wmma_bf16(ka1, qb1, s);
#pragma unroll
            for (int r = 0; r < 8; ++r)
                Ss[h][r + 8 * half][mt * 16 + lrow] = s[r] * (1.0f / 1024.0f);
        }
        __syncthreads();

        // softmax over the 8 heads, elementwise per (n,m)
#pragma unroll
        for (int e = 0; e < 2; ++e) {
            int idx = t + e * 256, n = idx >> 5, m = idx & 31;
            float v[NH], mx = -1e30f;
#pragma unroll
            for (int hh = 0; hh < NH; ++hh) { v[hh] = Ss[hh][n][m]; mx = fmaxf(mx, v[hh]); }
            float sum = 0.0f;
#pragma unroll
            for (int hh = 0; hh < NH; ++hh) { v[hh] = __expf(v[hh] - mx); sum += v[hh]; }
            float inv = 1.0f / sum;
#pragma unroll
            for (int hh = 0; hh < NH; ++hh) Ms[hh][n][m] = f2bf(v[hh] * inv);
        }
        __syncthreads();

        // att[n][c] += sm[n][m] * v[c][m]  (K = 32 m's in one WMMA)
        v16bf smA = load_frag_kmajor(&Ms[h][lrow][0], half);
#pragma unroll
        for (int cs = 0; cs < 4; ++cs) {
            v16bf vb = load_frag_kmajor(&Vs[h][cs * 16 + lrow][0], half);
            accv[cs] = wmma_bf16(smA, vb, accv[cs]);
        }
        __syncthreads();
    }

    // store att: [b][n][c*8+h]
#pragma unroll
    for (int cs = 0; cs < 4; ++cs)
#pragma unroll
        for (int r = 0; r < 8; ++r) {
            int n_l = r + 8 * half, c = cs * 16 + lrow;
            att[(size_t)b * NPIX * OC + (size_t)(nt + n_l) * OC + c * NH + h] =
                f2bf(accv[cs][r]);
        }
}

// ---------------------------------------------------------------------------
// K4: generic GEMM  out[row][col] = sum_k A[row][k] * Bw[col][k] + bias[col]
// double-buffered LDS with async staging.
// epilogue: 0 -> f32 store, 1 -> LeakyReLU(0.1) + bf16 store
// grid: (M/128, Ncols/64), block 256
// ---------------------------------------------------------------------------
__global__ __launch_bounds__(256) void gemm_kernel(const bf16_t* __restrict__ A,
                                                   const bf16_t* __restrict__ Bw,
                                                   const float* __restrict__ bias,
                                                   int K, int Ncols,
                                                   float* __restrict__ outF,
                                                   bf16_t* __restrict__ outB,
                                                   int epilogue) {
    __shared__ bf16_t As[2][128][32];
    __shared__ bf16_t Bs[2][64][32];
    const int t = threadIdx.x, wave = t >> 5, lane = t & 31;
    const int half = lane >> 4, lrow = lane & 15;
    const int rowblk = (wave & 3) * 32, colblk = (wave >> 2) * 32;
    const int mtile = blockIdx.x * 128, ctile = blockIdx.y * 64;

    const int a_r = t >> 1, a_seg = (t & 1) * 16;
    const int b_c = t >> 2, b_seg = (t & 3) * 8;

    auto loadA = [&](int k0, int buf) {
        const bf16_t* src = A + (size_t)(mtile + a_r) * K + k0 + a_seg;
        unsigned lo = (unsigned)(uintptr_t)&As[buf][a_r][a_seg];
        async_copy16(lo, src);
        async_copy16(lo + 16, src + 8);
    };
    auto loadB = [&](int k0, int buf) {
        const bf16_t* src = Bw + (size_t)(ctile + b_c) * K + k0 + b_seg;
        async_copy16((unsigned)(uintptr_t)&Bs[buf][b_c][b_seg], src);
    };

    v8f acc[2][2] = {};
    loadA(0, 0); loadB(0, 0);
    wait_async();
    __syncthreads();
    int cur = 0;
    for (int k0 = 0; k0 < K; k0 += 32) {
        int nxt = cur ^ 1;
        if (k0 + 32 < K) { loadA(k0 + 32, nxt); loadB(k0 + 32, nxt); }
        v16bf a0 = load_frag_kmajor(&As[cur][rowblk + lrow][0], half);
        v16bf a1 = load_frag_kmajor(&As[cur][rowblk + 16 + lrow][0], half);
        v16bf b0 = load_frag_kmajor(&Bs[cur][colblk + lrow][0], half);
        v16bf b1 = load_frag_kmajor(&Bs[cur][colblk + 16 + lrow][0], half);
        acc[0][0] = wmma_bf16(a0, b0, acc[0][0]);
        acc[0][1] = wmma_bf16(a0, b1, acc[0][1]);
        acc[1][0] = wmma_bf16(a1, b0, acc[1][0]);
        acc[1][1] = wmma_bf16(a1, b1, acc[1][1]);
        wait_async();
        __syncthreads();
        cur = nxt;
    }
#pragma unroll
    for (int i = 0; i < 2; ++i)
#pragma unroll
        for (int j = 0; j < 2; ++j)
#pragma unroll
            for (int r = 0; r < 8; ++r) {
                int row = mtile + rowblk + i * 16 + r + 8 * half;
                int col = ctile + colblk + j * 16 + lrow;
                float v = acc[i][j][r] + bias[col];
                if (epilogue == 1) {
                    v = v > 0.0f ? v : 0.1f * v;
                    outB[(size_t)row * Ncols + col] = f2bf(v);
                } else {
                    outF[(size_t)row * Ncols + col] = v;
                }
            }
}

// ---------------------------------------------------------------------------
// K5: add + LayerNorm over C=256. One block per row (b,n).
// res_t: residual read transposed from [B][C][N]; out_t: write transposed.
// ---------------------------------------------------------------------------
__global__ __launch_bounds__(256) void addln_kernel(const float* __restrict__ a,
                                                    const float* __restrict__ res,
                                                    int res_t,
                                                    const float* __restrict__ g,
                                                    const float* __restrict__ bta,
                                                    float* __restrict__ outF, int out_t,
                                                    bf16_t* __restrict__ outB) {
    __shared__ float red[256];
    const int t = threadIdx.x, row = blockIdx.x;
    const int b = row >> 10, n = row & (NPIX - 1);
    float rv = res_t ? res[(size_t)b * CDIM * NPIX + (size_t)t * NPIX + n]
                     : res[(size_t)row * CDIM + t];
    float v = a[(size_t)row * CDIM + t] + rv;

    red[t] = v; __syncthreads();
    for (int s = 128; s > 0; s >>= 1) { if (t < s) red[t] += red[t + s]; __syncthreads(); }
    float mu = red[0] * (1.0f / 256.0f);
    __syncthreads();
    float d = v - mu;
    red[t] = d * d; __syncthreads();
    for (int s = 128; s > 0; s >>= 1) { if (t < s) red[t] += red[t + s]; __syncthreads(); }
    float var = red[0] * (1.0f / 256.0f);
    float y = d * rsqrtf(var + 1e-5f) * g[t] + bta[t];

    if (out_t)
        outF[(size_t)b * CDIM * NPIX + (size_t)t * NPIX + n] = y;
    else
        outF[(size_t)row * CDIM + t] = y;
    if (outB) outB[(size_t)row * CDIM + t] = f2bf(y);
}

// ---------------------------------------------------------------------------
// launch
// ---------------------------------------------------------------------------
extern "C" void kernel_launch(void* const* d_in, const int* in_sizes, int n_in,
                              void* d_out, int out_size, void* d_ws, size_t ws_size,
                              hipStream_t stream) {
    const float* x     = (const float*)d_in[0];
    const float* Wk    = (const float*)d_in[1];
    const float* bk    = (const float*)d_in[2];
    const float* Wq    = (const float*)d_in[3];
    const float* bq    = (const float*)d_in[4];
    const float* Wv    = (const float*)d_in[5];
    const float* bv    = (const float*)d_in[6];
    const float* Wproj = (const float*)d_in[7];
    const float* bproj = (const float*)d_in[8];
    const float* ln_g  = (const float*)d_in[9];
    const float* ln_b  = (const float*)d_in[10];
    const float* W1    = (const float*)d_in[11];
    const float* b1    = (const float*)d_in[12];
    const float* W2    = (const float*)d_in[13];
    const float* b2    = (const float*)d_in[14];

    char* ws = (char*)d_ws;
    size_t off = 0;
    auto alloc = [&](size_t bytes) { size_t p = off; off += (bytes + 255) & ~(size_t)255; return p; };
    const size_t ACT  = (size_t)BDIM * CDIM * NPIX;      // 2M elems
    const size_t KQVN = (size_t)BDIM * OC * NPIX;        // 4M elems

    float*  xpF    = (float*) (ws + alloc(ACT * 4));
    bf16_t* xpB    = (bf16_t*)(ws + alloc(ACT * 2));
    bf16_t* WkB    = (bf16_t*)(ws + alloc((size_t)OC * KIM * 2));
    bf16_t* WqB    = (bf16_t*)(ws + alloc((size_t)OC * KIM * 2));
    bf16_t* WvB    = (bf16_t*)(ws + alloc((size_t)OC * KIM * 2));
    bf16_t* WpB    = (bf16_t*)(ws + alloc((size_t)CDIM * OC * 2));
    bf16_t* W1B    = (bf16_t*)(ws + alloc((size_t)CDIM * CDIM * 2));
    bf16_t* W2B    = (bf16_t*)(ws + alloc((size_t)CDIM * CDIM * 2));
    bf16_t* kB     = (bf16_t*)(ws + alloc(KQVN * 2));
    bf16_t* qB     = (bf16_t*)(ws + alloc(KQVN * 2));
    bf16_t* vB     = (bf16_t*)(ws + alloc(KQVN * 2));
    bf16_t* attB   = (bf16_t*)(ws + alloc(KQVN * 2));
    float*  aprojF = (float*) (ws + alloc(ACT * 4));
    float*  anF    = (float*) (ws + alloc(ACT * 4));
    bf16_t* anB    = (bf16_t*)(ws + alloc(ACT * 2));
    bf16_t* h1B    = (bf16_t*)(ws + alloc(ACT * 2));
    float*  h2F    = (float*) (ws + alloc(ACT * 4));
    (void)ws_size; (void)n_in; (void)in_sizes; (void)out_size;

    // conv weight permute+convert (tap-major), linear converts for the rest
    int nW = OC * KIM;
    permute_wconv_kernel<<<(nW + 255) / 256, 256, 0, stream>>>(Wk, WkB);
    permute_wconv_kernel<<<(nW + 255) / 256, 256, 0, stream>>>(Wq, WqB);
    permute_wconv_kernel<<<(nW + 255) / 256, 256, 0, stream>>>(Wv, WvB);
    convert_kernel<<<(CDIM * OC + 255) / 256, 256, 0, stream>>>(Wproj, WpB, CDIM * OC);
    convert_kernel<<<(CDIM * CDIM + 255) / 256, 256, 0, stream>>>(W1, W1B, CDIM * CDIM);
    convert_kernel<<<(CDIM * CDIM + 255) / 256, 256, 0, stream>>>(W2, W2B, CDIM * CDIM);

    // xp = x + pe
    pe_add_kernel<<<(int)(ACT / 256), 256, 0, stream>>>(x, xpF, xpB);

    // q/k/v convs
    dim3 cgrid(NPIX / 64, OC / 128, BDIM);
    conv_gemm_kernel<<<cgrid, 256, 0, stream>>>(WkB, bk, xpB, kB);
    conv_gemm_kernel<<<cgrid, 256, 0, stream>>>(WqB, bq, xpB, qB);
    conv_gemm_kernel<<<cgrid, 256, 0, stream>>>(WvB, bv, xpB, vB);

    // fused attention
    attention_kernel<<<dim3(NPIX / 16, BDIM), 256, 0, stream>>>(kB, qB, vB, attB);

    // projection: [8192,512] x [256,512]^T -> f32
    gemm_kernel<<<dim3(BDIM * NPIX / 128, CDIM / 64), 256, 0, stream>>>(
        attB, WpB, bproj, OC, CDIM, aprojF, nullptr, 0);

    // addnorm1 = LN(xflat + attproj)
    addln_kernel<<<BDIM * NPIX, 256, 0, stream>>>(aprojF, xpF, 1, ln_g, ln_b,
                                                  anF, 0, anB);

    // FFN
    gemm_kernel<<<dim3(BDIM * NPIX / 128, CDIM / 64), 256, 0, stream>>>(
        anB, W1B, b1, CDIM, CDIM, nullptr, h1B, 1);
    gemm_kernel<<<dim3(BDIM * NPIX / 128, CDIM / 64), 256, 0, stream>>>(
        h1B, W2B, b2, CDIM, CDIM, h2F, nullptr, 0);

    // out = LN(h2 + addnorm1), written transposed to [B,C,H,W]
    addln_kernel<<<BDIM * NPIX, 256, 0, stream>>>(h2F, anF, 0, ln_g, ln_b,
                                                  (float*)d_out, 1, nullptr);
}
